// ScaledDotProductAttention_50921132261424
// MI455X (gfx1250) — matrix operand order repaired and checked by reading, not compiled, not run on hardware
//
#include <hip/hip_runtime.h>
#include <cmath>

typedef __attribute__((ext_vector_type(16))) _Float16 v16h;
typedef __attribute__((ext_vector_type(8)))  float    v8f;
typedef __attribute__((ext_vector_type(4)))  float    v4f;   // native clang vector for NT stores

#define WAVES  8
#define BLOCKT (WAVES * 32)

// Build an A-fragment (16x32 f16, M = lane%16) from 32 consecutive f32 at `src`,
// multiplying by `scl` before conversion.
// Layout: base = (lane>=16)?16:0 ; a[j]=src[base+j]*scl for j=0..15.
__device__ __forceinline__ v16h load_afrag_f32row_scaled(const float* __restrict__ src,
                                                         int base, float scl) {
    const float4 p0 = *(const float4*)(src + base);
    const float4 p1 = *(const float4*)(src + base + 4);
    const float4 p2 = *(const float4*)(src + base + 8);
    const float4 p3 = *(const float4*)(src + base + 12);
    v16h a;
    a[0]  = (_Float16)(p0.x * scl); a[1]  = (_Float16)(p0.y * scl);
    a[2]  = (_Float16)(p0.z * scl); a[3]  = (_Float16)(p0.w * scl);
    a[4]  = (_Float16)(p1.x * scl); a[5]  = (_Float16)(p1.y * scl);
    a[6]  = (_Float16)(p1.z * scl); a[7]  = (_Float16)(p1.w * scl);
    a[8]  = (_Float16)(p2.x * scl); a[9]  = (_Float16)(p2.y * scl);
    a[10] = (_Float16)(p2.z * scl); a[11] = (_Float16)(p2.w * scl);
    a[12] = (_Float16)(p3.x * scl); a[13] = (_Float16)(p3.y * scl);
    a[14] = (_Float16)(p3.z * scl); a[15] = (_Float16)(p3.w * scl);
    return a;
}

// Build an A-fragment (16x32 f16, M = lane%16) from 32 consecutive f32 at `src`.
__device__ __forceinline__ v16h load_afrag_f32row(const float* __restrict__ src, int base) {
    const float4 p0 = *(const float4*)(src + base);
    const float4 p1 = *(const float4*)(src + base + 4);
    const float4 p2 = *(const float4*)(src + base + 8);
    const float4 p3 = *(const float4*)(src + base + 12);
    v16h a;
    a[0]  = (_Float16)p0.x; a[1]  = (_Float16)p0.y; a[2]  = (_Float16)p0.z; a[3]  = (_Float16)p0.w;
    a[4]  = (_Float16)p1.x; a[5]  = (_Float16)p1.y; a[6]  = (_Float16)p1.z; a[7]  = (_Float16)p1.w;
    a[8]  = (_Float16)p2.x; a[9]  = (_Float16)p2.y; a[10] = (_Float16)p2.z; a[11] = (_Float16)p2.w;
    a[12] = (_Float16)p3.x; a[13] = (_Float16)p3.y; a[14] = (_Float16)p3.z; a[15] = (_Float16)p3.w;
    return a;
}

// Build a B-fragment (32x16 f16, N = lane%16) from 16 consecutive f32 at `src`
// (this lane's column data along K, already offset by k0 + koff, koff=(lane>=16)?16:0).
__device__ __forceinline__ v16h load_bfrag_f32row(const float* __restrict__ src) {
    const float4 p0 = *(const float4*)(src + 0);
    const float4 p1 = *(const float4*)(src + 4);
    const float4 p2 = *(const float4*)(src + 8);
    const float4 p3 = *(const float4*)(src + 12);
    v16h b;
    b[0]  = (_Float16)p0.x; b[1]  = (_Float16)p0.y; b[2]  = (_Float16)p0.z; b[3]  = (_Float16)p0.w;
    b[4]  = (_Float16)p1.x; b[5]  = (_Float16)p1.y; b[6]  = (_Float16)p1.z; b[7]  = (_Float16)p1.w;
    b[8]  = (_Float16)p2.x; b[9]  = (_Float16)p2.y; b[10] = (_Float16)p2.z; b[11] = (_Float16)p2.w;
    b[12] = (_Float16)p3.x; b[13] = (_Float16)p3.y; b[14] = (_Float16)p3.z; b[15] = (_Float16)p3.w;
    return b;
}

__global__ __launch_bounds__(BLOCKT)
void sdpa_fwd_kernel(const float* __restrict__ q,
                     const float* __restrict__ k,
                     const float* __restrict__ v,
                     const float* __restrict__ mask,
                     float* __restrict__ out,       // [B,S,D]
                     float* __restrict__ attn,      // [B,S,S]
                     int S, float scale) {
    constexpr int D = 128;
    extern __shared__ float smem[];
    const int SP = S + 4;                 // padded LDS row stride (dwords)
    float* sP     = smem;                 // 16 x SP score/prob buffer
    float* rowInv = smem + 16 * SP;       // 16 floats

    const int tid     = threadIdx.x;
    const int wave    = tid >> 5;
    const int lane    = tid & 31;
    const int m       = lane & 15;        // fragment row/col index
    const int halfSel = lane >> 4;
    const int baseA   = halfSel * 16;     // A-fragment K sub-offset
    const int koffB   = halfSel * 16;     // B-fragment K sub-offset

    const int b  = blockIdx.y;
    const int q0 = blockIdx.x * 16;

    const float* __restrict__ Qb = q + ((size_t)b * S + q0) * D;
    const float* __restrict__ Kb = k + (size_t)b * S * D;
    const float* __restrict__ Vb = v + (size_t)b * S * D;

    // ---- Q fragments (scale folded into the f16 conversion) ----
    v16h qf[4];
    {
        const float* qrow = Qb + m * D;   // lane's M row
        #pragma unroll
        for (int c = 0; c < 4; ++c)
            qf[c] = load_afrag_f32row_scaled(qrow + c * 32, baseA, scale);
    }

    // ---- Phase 1: scores = (Q*scale) K^T + mask, staged in LDS ----
    const int nKT = S >> 4;               // key tiles of 16
    for (int kt = wave; kt < nKT; kt += WAVES) {
        const int krow = kt * 16 + m;     // this lane's key row (B-frag column N=m)
        const float* krp = Kb + (size_t)krow * D + koffB;
        v8f acc = {};
        #pragma unroll
        for (int c = 0; c < 4; ++c) {
            v16h bf = load_bfrag_f32row(krp + c * 32);
            acc = __builtin_amdgcn_wmma_f32_16x16x32_f16(
                false, qf[c], false, bf, (short)0, acc, false, false);
        }
        const int col = kt * 16 + m;
        #pragma unroll
        for (int r = 0; r < 8; ++r) {
            const int M = r + halfSel * 8;
            sP[M * SP + col] = acc[r] + mask[(size_t)(q0 + M) * S + col];
        }
    }
    __syncthreads();

    // ---- Phase 2: per-row softmax, float4 sweeps; each wave owns full rows ----
    const int n4 = S >> 2;
    for (int row = wave; row < 16; row += WAVES) {
        v4f* prow4 = (v4f*)(sP + row * SP);

        float rmax = -INFINITY;
        for (int i = lane; i < n4; i += 32) {
            v4f t = prow4[i];
            rmax = fmaxf(rmax, fmaxf(fmaxf(t.x, t.y), fmaxf(t.z, t.w)));
        }
        #pragma unroll
        for (int off = 16; off > 0; off >>= 1)
            rmax = fmaxf(rmax, __shfl_xor(rmax, off, 32));

        float sum = 0.0f;
        for (int i = lane; i < n4; i += 32) {
            v4f t = prow4[i];
            t.x = __expf(t.x - rmax); t.y = __expf(t.y - rmax);
            t.z = __expf(t.z - rmax); t.w = __expf(t.w - rmax);
            prow4[i] = t;                 // keep unnormalized exp in LDS
            sum += (t.x + t.y) + (t.z + t.w);
        }
        #pragma unroll
        for (int off = 16; off > 0; off >>= 1)
            sum += __shfl_xor(sum, off, 32);

        const float inv = 1.0f / sum;
        if (lane == 0) rowInv[row] = inv;

        // Streaming, write-once: non-temporal b128 stores of normalized probs.
        v4f* arow4 = (v4f*)(attn + ((size_t)b * S + (q0 + row)) * S);
        for (int i = lane; i < n4; i += 32) {
            v4f t = prow4[i] * inv;       // ext-vector splat multiply
            __builtin_nontemporal_store(t, arow4 + i);
        }
    }
    __syncthreads();

    // ---- Phase 3: out = softmax(P) * V ; wave owns 16 output columns ----
    const int colBase = wave * 16;        // WAVES*16 == D == 128
    const int colV    = colBase + m;      // B-frag column (N=m)
    v8f oacc = {};
    for (int k0 = 0; k0 < S; k0 += 32) {
        // A: probability fragment from LDS (unnormalized exp)
        v16h af = load_afrag_f32row(sP + m * SP + k0, baseA);
        // B: V fragment, b[e] = V[k0+koffB+e][colV] (L2-resident slice)
        v16h bf;
        const float* vp = Vb + (size_t)(k0 + koffB) * D + colV;
        #pragma unroll
        for (int e = 0; e < 16; ++e)
            bf[e] = (_Float16)vp[(size_t)e * D];
        oacc = __builtin_amdgcn_wmma_f32_16x16x32_f16(
            false, af, false, bf, (short)0, oacc, false, false);
    }
    #pragma unroll
    for (int r = 0; r < 8; ++r) {
        const int M = r + halfSel * 8;    // fold 1/rowsum into the epilogue
        __builtin_nontemporal_store(oacc[r] * rowInv[M],
            &out[((size_t)b * S + (q0 + M)) * D + colBase + m]);
    }
}

extern "C" void kernel_launch(void* const* d_in, const int* in_sizes, int n_in,
                              void* d_out, int out_size, void* d_ws, size_t ws_size,
                              hipStream_t stream) {
    (void)n_in; (void)d_ws; (void)ws_size; (void)out_size;
    const float* q    = (const float*)d_in[0];
    const float* k    = (const float*)d_in[1];
    const float* v    = (const float*)d_in[2];
    const float* mask = (const float*)d_in[3];

    const int D = 128;
    int S = (int)(sqrt((double)in_sizes[3]) + 0.5);   // mask is [S,S]
    int B = in_sizes[0] / (S * D);

    float* out  = (float*)d_out;                       // [B,S,D] first
    float* attn = out + (size_t)B * S * D;             // then [B,S,S]

    const float scale = 1.0f / sqrtf((float)D);
    dim3 grid(S / 16, B);
    dim3 block(BLOCKT);
    size_t shmem = (size_t)(16 * (S + 4) + 16) * sizeof(float);

    sdpa_fwd_kernel<<<grid, block, shmem, stream>>>(q, k, v, mask, out, attn, S, scale);
}
